// Cybond_Punk_Network_77197742178846
// MI455X (gfx1250) — matrix operand order repaired and checked by reading, not compiled, not run on hardware
//
#include <hip/hip_runtime.h>
#include <hip/hip_bf16.h>

typedef __bf16 bf16_t;
typedef __attribute__((ext_vector_type(16))) __bf16 v16bf;
typedef __attribute__((ext_vector_type(8)))  __bf16 v8bf;
typedef __attribute__((ext_vector_type(8)))  float  v8f;

static const int NN  = 20000;
static const int EE  = 640000;
static const int MM  = 200000;
static const int NCY = 50000;
static const int GG  = 512;
static const float EPSV = 1e-5f;

// ---------------------------------------------------------------------------
// CDNA5 async global->LDS copy (per-lane LDS dest address, ASYNCcnt-tracked).
// Generic pointers to __shared__ carry the LDS byte offset in their low 32
// bits (flat aperture rule), which is exactly what the VDST VGPR wants.
// ---------------------------------------------------------------------------
__device__ __forceinline__ void async_g2l_b128(void* ldsp, const void* gp) {
  unsigned off = (unsigned)(size_t)ldsp;
  asm volatile("global_load_async_to_lds_b128 %0, %1, off"
               :: "v"(off), "v"(gp) : "memory");
}
__device__ __forceinline__ void wait_async0() {
  asm volatile("s_wait_asynccnt 0x0" ::: "memory");
}

// ---------------------------------------------------------------------------
// Software-pipelined WMMA pass: 16x128 output slab per wave, K=128.
// A/B fragments are stored in LDS per lane in k = 16*half + slot order, so each
// fragment is one aligned 32B load (2x ds_load_b128). The next fragment's
// loads are issued before the current WMMA to hide LDS latency.
// ---------------------------------------------------------------------------
__device__ __forceinline__ void gemm_pass(const v16bf* __restrict__ Af,
                                          const v16bf* __restrict__ Bf,
                                          v8f* acc, int lane, int wave)
{
  v16bf a = Af[(wave * 4 + 0) * 32 + lane];
  v16bf b = Bf[(0 * 8 + 0) * 32 + lane];
#pragma unroll
  for (int t = 0; t < 32; ++t) {
    const int nt = t & 7;
    v16bf an = a, bn = b;
    if (t < 31) {
      const int t2 = t + 1;
      const int kb2 = t2 >> 3, nt2 = t2 & 7;
      bn = Bf[(kb2 * 8 + nt2) * 32 + lane];
      if (nt2 == 0) an = Af[(wave * 4 + kb2) * 32 + lane];
    }
    acc[nt] = __builtin_amdgcn_wmma_f32_16x16x32_bf16(
        false, a, false, b, (short)0, acc[nt], false, false);
    a = an; b = bn;
  }
}

// Chain GEMM computed transposed: D1 = W1^T x X^T ; D2 = W2^T x D1.
// Y[row][col] = D2[col][row] (+ biases), per-column sum/sumsq for BN.
// MODE 0: X bf16 rows x 128 (async global->LDS staging).
// MODE 1: edge concat gather [x_bf[src]|x_bf[dst]]. MODE 2: X f32 rows x 128.
template<int MODE>
__global__ __launch_bounds__(256)
void k_gemm_chain(const void* __restrict__ Xv,
                  const int* __restrict__ srcIdx, const int* __restrict__ dstIdx,
                  const float* __restrict__ W1, const float* __restrict__ B1,
                  const float* __restrict__ W2, const float* __restrict__ B2,
                  float* __restrict__ Y, int rows,
                  float* __restrict__ colsum, float* __restrict__ colsq)
{
  __shared__ alignas(32) bf16_t sWA1[128 * 128];  // W1^T, A-frag layout
  __shared__ alignas(32) bf16_t sWA2[128 * 128];  // W2^T, A-frag layout
  __shared__ alignas(32) bf16_t sB  [128 * 128];  // X^T then T, B-frag layout
  __shared__ float sStat[256];

  const int tid = threadIdx.x;

  // ---- stage transposed weights into A-fragment layout ----
  for (int idx = tid; idx < 128 * 128; idx += 256) {
    const int n = idx & 127;          // output feature (row of W^T)
    const int k = idx >> 7;           // contraction dim (W row)
    const int kk = k & 31;
    const int halfk = kk >> 4;
    const int slot  = kk & 15;
    const int offA  = (((n >> 4) * 4 + (k >> 5)) * 32 + halfk * 16 + (n & 15)) * 16 + slot;
    sWA1[offA] = (bf16_t)W1[idx];     // idx == k*128 + n
    sWA2[offA] = (bf16_t)W2[idx];
  }
  sStat[tid] = 0.0f;

  // ---- stage input tile into B-fragment layout: B[k][m] = X[m][k] ----
  const int tile0 = blockIdx.x * 128;
  const int tr = tid >> 1;            // tile row (m) this thread fills
  const int th = tid & 1;             // which 64-k half
  const int grow = tile0 + tr;
  {
    const int mwt = tr >> 4, mr = tr & 15;
#pragma unroll
    for (int c = 0; c < 4; ++c) {
      const int k0 = th * 64 + c * 16;
      const int chunk = ((k0 >> 5) * 8 + mwt) * 32 + ((k0 & 31) >> 4) * 16 + mr;
      uint4* d4 = (uint4*)&sB[chunk * 16];
      if (grow < rows) {
        if (MODE == 1) {
          const int* idxp = th ? dstIdx : srcIdx;
          __builtin_prefetch(idxp + grow + 128, 0, 0);
          const int node = idxp[grow];
          const uint4* s4 = (const uint4*)((const bf16_t*)Xv + (size_t)node * 64 + c * 16);
          d4[0] = s4[0]; d4[1] = s4[1];
        } else if (MODE == 0) {
          const bf16_t* sp = (const bf16_t*)Xv + (size_t)grow * 128 + k0;
          async_g2l_b128(&sB[chunk * 16],     sp);
          async_g2l_b128(&sB[chunk * 16 + 8], sp + 8);
        } else {
          const float* sp = (const float*)Xv + (size_t)grow * 128 + k0;
          v8bf p0, p1;
#pragma unroll
          for (int j = 0; j < 8; ++j) { p0[j] = (bf16_t)sp[j]; p1[j] = (bf16_t)sp[8 + j]; }
          ((v8bf*)d4)[0] = p0; ((v8bf*)d4)[1] = p1;
        }
      } else {
        const uint4 z = {0u, 0u, 0u, 0u};
        d4[0] = z; d4[1] = z;
      }
    }
  }
  if (MODE == 0) wait_async0();
  __syncthreads();

  const int lane = tid & 31;
  const int wave = tid >> 5;          // A row-tile: features [wave*16, +16)
  const int half = lane >> 4;
  const int r    = lane & 15;
  const int nb   = wave * 16 + half * 8;   // this thread's 8 feature columns

  const v8f zf = {0.f, 0.f, 0.f, 0.f, 0.f, 0.f, 0.f, 0.f};
  v8f acc[8];
#pragma unroll
  for (int nt = 0; nt < 8; ++nt) acc[nt] = zf;

  const v16bf* Af1 = (const v16bf*)sWA1;
  const v16bf* Af2 = (const v16bf*)sWA2;
  const v16bf* Bf  = (const v16bf*)sB;

  // ---- GEMM1': D1 = W1^T x X^T ----
  gemm_pass(Af1, Bf, acc, lane, wave);
  __syncthreads();                    // everyone done reading sB

  // ---- T = D1 + b1 -> sB in B-fragment layout (k = feature, n = row m) ----
  {
    float bias1[8];
#pragma unroll
    for (int j = 0; j < 8; ++j) bias1[j] = B1 ? B1[nb + j] : 0.0f;
    const int c5 = nb >> 5, h2 = (nb & 31) >> 4, s0 = nb & 15;
#pragma unroll
    for (int nt = 0; nt < 8; ++nt) {
      v8bf pk;
#pragma unroll
      for (int j = 0; j < 8; ++j) pk[j] = (bf16_t)(acc[nt][j] + bias1[j]);
      const int chunk = (c5 * 8 + nt) * 32 + h2 * 16 + r;
      *(v8bf*)&sB[chunk * 16 + s0] = pk;     // one aligned 16B LDS store
    }
  }
  __syncthreads();

  // ---- GEMM2': D2 = W2^T x T ----
#pragma unroll
  for (int nt = 0; nt < 8; ++nt) acc[nt] = zf;
  gemm_pass(Af2, Bf, acc, lane, wave);

  // ---- epilogue: Y[row][col] = D2[col][row] + b2, BN column stats ----
  {
    float bias2[8];
#pragma unroll
    for (int j = 0; j < 8; ++j) bias2[j] = B2 ? B2[nb + j] : 0.0f;
    float s[8] = {0.f, 0.f, 0.f, 0.f, 0.f, 0.f, 0.f, 0.f};
    float q[8] = {0.f, 0.f, 0.f, 0.f, 0.f, 0.f, 0.f, 0.f};
#pragma unroll
    for (int nt = 0; nt < 8; ++nt) {
      const int gr = tile0 + nt * 16 + r;
      if (gr < rows) {
        float* yp = Y + (size_t)gr * 128 + nb;
#pragma unroll
        for (int j = 0; j < 8; ++j) {
          const float v = acc[nt][j] + bias2[j];
          yp[j] = v;                 // 8 consecutive floats -> wide store
          s[j] += v;
          q[j] += v * v;
        }
      }
    }
    if (colsum != nullptr) {
#pragma unroll
      for (int j = 0; j < 8; ++j) {
        atomicAdd(&sStat[nb + j], s[j]);
        atomicAdd(&sStat[128 + nb + j], q[j]);
      }
    }
  }
  __syncthreads();
  if (colsum != nullptr && tid < 128) {
    atomicAdd(&colsum[tid], sStat[tid]);
    atomicAdd(&colsq[tid],  sStat[128 + tid]);
  }
}

// ---------------------------------------------------------------------------
// Elementwise / reduction helper kernels
// ---------------------------------------------------------------------------
__global__ void k_zero(float* p, long long n) {
  long long i = (long long)blockIdx.x * blockDim.x + threadIdx.x;
  if (i < n) p[i] = 0.0f;
}

__global__ void k_embed(const int* sub_x, const float* emb, bf16_t* x_bf, long long n) {
  long long i = (long long)blockIdx.x * blockDim.x + threadIdx.x;
  if (i >= n) return;
  int node = (int)(i >> 6), c = (int)(i & 63);
  x_bf[i] = (bf16_t)emb[sub_x[node] * 64 + c];
}

__global__ void k_bn_fin(const float* colsum, const float* colsq, float rowsInv,
                         float* meanInv) {
  int c = threadIdx.x;
  if (c < 128) {
    float m = colsum[c] * rowsInv;
    float v = colsq[c] * rowsInv - m * m;
    meanInv[c] = m;
    meanInv[128 + c] = rsqrtf(v + EPSV);
  }
}

__global__ void k_bn_act(const float* __restrict__ Y, const float* __restrict__ meanInv,
                         const float* __restrict__ g, const float* __restrict__ beta,
                         float* __restrict__ outF, bf16_t* __restrict__ outBf,
                         long long n) {
  long long i = (long long)blockIdx.x * blockDim.x + threadIdx.x;
  if (i >= n) return;
  int c = (int)(i & 127);
  float y = g[c] * (Y[i] - meanInv[c]) * meanInv[128 + c] + beta[c];
  float ge = 0.5f * y * (1.0f + erff(y * 0.70710678118654752f));
  if (outF)  outF[i]  = ge;
  if (outBf) outBf[i] = (bf16_t)ge;
}

// per-row attention logit: leaky_relu(dot(h_row, watt)); one wave per row
__global__ __launch_bounds__(256)
void k_att(const bf16_t* __restrict__ H, const int* __restrict__ rowIdx,
           const float* __restrict__ watt, float* __restrict__ att, int rows) {
  int row = blockIdx.x * 8 + (threadIdx.x >> 5);
  if (row >= rows) return;
  int lane = threadIdx.x & 31;
  long long src = rowIdx ? (long long)rowIdx[row] : (long long)row;
  const bf16_t* hr = H + src * 128;
  float s = 0.0f;
#pragma unroll
  for (int i = 0; i < 4; ++i) {
    int c = lane * 4 + i;
    s += (float)hr[c] * watt[c];
  }
  for (int off = 16; off; off >>= 1) s += __shfl_xor(s, off, 32);
  if (lane == 0) att[row] = (s > 0.0f) ? s : 0.01f * s;
}

__device__ __forceinline__ unsigned f_enc(float f) {
  unsigned u = __float_as_uint(f);
  return (u & 0x80000000u) ? ~u : (u | 0x80000000u);
}
__device__ __forceinline__ float f_dec(unsigned u) {
  return (u & 0x80000000u) ? __uint_as_float(u & 0x7FFFFFFFu)
                           : __uint_as_float(~u);
}

__global__ void k_seg_max(const float* att, const int* seg, unsigned* umax, int n) {
  int i = blockIdx.x * blockDim.x + threadIdx.x;
  if (i < n) atomicMax(&umax[seg[i]], f_enc(att[i]));
}

__global__ void k_seg_expsum(const float* att, const int* seg, const unsigned* umax,
                             float* ex, float* ssum, float* cnt, int n) {
  int i = blockIdx.x * blockDim.x + threadIdx.x;
  if (i >= n) return;
  int s = seg[i];
  float e = expf(att[i] - f_dec(umax[s]));
  ex[i] = e;
  atomicAdd(&ssum[s], e);
  atomicAdd(&cnt[s], 1.0f);
}

__global__ void k_seg_acc(const bf16_t* __restrict__ H, const int* __restrict__ rowIdx,
                          const int* __restrict__ seg, const float* __restrict__ ex,
                          const float* __restrict__ ssum, float* __restrict__ acc,
                          long long n) {
  long long i = (long long)blockIdx.x * blockDim.x + threadIdx.x;
  if (i >= n) return;
  int e = (int)(i >> 7), c = (int)(i & 127);
  int s = seg[e];
  float w = ex[e] / ssum[s];
  long long hrow = rowIdx ? (long long)rowIdx[e] : (long long)e;
  atomicAdd(&acc[(long long)s * 128 + c], w * (float)H[hrow * 128 + c]);
}

__global__ void k_div_cnt(const float* acc, const float* cnt, bf16_t* out, long long n) {
  long long i = (long long)blockIdx.x * blockDim.x + threadIdx.x;
  if (i >= n) return;
  out[i] = (bf16_t)(acc[i] / fmaxf(cnt[i >> 7], 1.0f));
}

__global__ void k_scatter_add(const int* __restrict__ rowIdx, const int* __restrict__ seg,
                              const float* __restrict__ srcM, float* __restrict__ dstM,
                              long long n) {
  long long i = (long long)blockIdx.x * blockDim.x + threadIdx.x;
  if (i >= n) return;
  int m = (int)(i >> 7), c = (int)(i & 127);
  atomicAdd(&dstM[(long long)rowIdx[m] * 128 + c],
            srcM[(long long)seg[m] * 128 + c]);
}

__global__ void k_add_bond(float* __restrict__ e, const float* __restrict__ bondF,
                           const int* __restrict__ src, const int* __restrict__ dst,
                           long long n) {
  long long i = (long long)blockIdx.x * blockDim.x + threadIdx.x;
  if (i >= n) return;
  int ed = (int)(i >> 7), c = (int)(i & 127);
  e[i] += bondF[(long long)src[ed] * 128 + c] + bondF[(long long)dst[ed] * 128 + c];
}

__global__ void k_graph_sum(const float* __restrict__ e2, const int* __restrict__ src,
                            const int* __restrict__ batch, float* __restrict__ gsum,
                            long long n) {
  long long i = (long long)blockIdx.x * blockDim.x + threadIdx.x;
  if (i >= n) return;
  int ed = (int)(i >> 7), c = (int)(i & 127);
  atomicAdd(&gsum[(long long)batch[src[ed]] * 128 + c], e2[i]);
}

__global__ void k_out(const float* __restrict__ gF, const float* __restrict__ wout,
                      float* __restrict__ out, int gcount) {
  int i = blockIdx.x * blockDim.x + threadIdx.x;
  if (i >= gcount * 2) return;
  int g = i >> 1, k = i & 1;
  float s = 0.0f;
  for (int c = 0; c < 128; ++c) s += gF[g * 128 + c] * wout[c * 2 + k];
  out[i] = s;
}

// ---------------------------------------------------------------------------
// Host orchestration
// ---------------------------------------------------------------------------
extern "C" void kernel_launch(void* const* d_in, const int* in_sizes, int n_in,
                              void* d_out, int out_size, void* d_ws, size_t ws_size,
                              hipStream_t stream) {
  (void)in_sizes; (void)n_in; (void)out_size; (void)ws_size;

  const int*   sub_x  = (const int*)d_in[0];
  const int*   eidx   = (const int*)d_in[1];
  const int*   src    = eidx;
  const int*   dst    = eidx + EE;
  const int*   batch  = (const int*)d_in[2];
  const int*   cseg   = (const int*)d_in[3];
  const int*   cidx   = (const int*)d_in[4];
  const float* emb    = (const float*)d_in[5];
  const float* w1     = (const float*)d_in[6];
  const float* b1     = (const float*)d_in[7];
  const float* w2     = (const float*)d_in[8];
  const float* b2     = (const float*)d_in[9];
  const float* g0     = (const float*)d_in[10];
  const float* beta0  = (const float*)d_in[11];
  const float* watt_b = (const float*)d_in[12];
  const float* wb1    = (const float*)d_in[13];
  const float* wb2    = (const float*)d_in[14];
  const float* g1     = (const float*)d_in[15];
  const float* beta1  = (const float*)d_in[16];
  const float* watt_c = (const float*)d_in[17];
  const float* wc1    = (const float*)d_in[18];
  const float* wc2    = (const float*)d_in[19];
  const float* g2     = (const float*)d_in[20];
  const float* beta2  = (const float*)d_in[21];
  const float* we1    = (const float*)d_in[22];
  const float* we2    = (const float*)d_in[23];
  const float* wl1    = (const float*)d_in[24];
  const float* wl2    = (const float*)d_in[25];
  const float* g3     = (const float*)d_in[26];
  const float* beta3  = (const float*)d_in[27];
  const float* wg1    = (const float*)d_in[28];
  const float* wg2    = (const float*)d_in[29];
  const float* g4     = (const float*)d_in[30];
  const float* beta4  = (const float*)d_in[31];
  const float* wout   = (const float*)d_in[32];

  char* w = (char*)d_ws;
  size_t off = 0;
  auto alloc = [&](size_t bytes) -> void* {
    void* p = w + off;
    off += (bytes + 255) & ~(size_t)255;
    return p;
  };

  float*    bufA    = (float*)   alloc((size_t)EE * 128 * 4);
  bf16_t*   h_bf    = (bf16_t*)  alloc((size_t)EE * 128 * 2);
  float*    e_buf   = (float*)   alloc((size_t)EE * 128 * 4);
  float*    att_e   = (float*)   alloc((size_t)EE * 4);
  float*    ex_e    = (float*)   alloc((size_t)EE * 4);
  unsigned* umaxN   = (unsigned*)alloc((size_t)NN * 4);
  float*    ssumN   = (float*)   alloc((size_t)NN * 4);
  float*    cntN    = (float*)   alloc((size_t)NN * 4);
  float*    accN    = (float*)   alloc((size_t)NN * 128 * 4);
  bf16_t*   bond_bf = (bf16_t*)  alloc((size_t)NN * 128 * 2);
  float*    Ybond   = (float*)   alloc((size_t)NN * 128 * 4);
  float*    bondF   = (float*)   alloc((size_t)NN * 128 * 4);
  float*    att_c   = (float*)   alloc((size_t)MM * 4);
  float*    ex_c    = (float*)   alloc((size_t)MM * 4);
  unsigned* umaxC   = (unsigned*)alloc((size_t)NCY * 4);
  float*    ssumC   = (float*)   alloc((size_t)NCY * 4);
  float*    cntC    = (float*)   alloc((size_t)NCY * 4);
  float*    accC    = (float*)   alloc((size_t)NCY * 128 * 4);
  bf16_t*   cyc_bf  = (bf16_t*)  alloc((size_t)NCY * 128 * 2);
  float*    Ycyc    = (float*)   alloc((size_t)NCY * 128 * 4);
  float*    cycF    = (float*)   alloc((size_t)NCY * 128 * 4);
  float*    gsum    = (float*)   alloc((size_t)GG * 128 * 4);
  float*    Yg      = (float*)   alloc((size_t)GG * 128 * 4);
  float*    gF      = (float*)   alloc((size_t)GG * 128 * 4);
  bf16_t*   x_bf    = (bf16_t*)  alloc((size_t)NN * 64 * 2);
  float*    stats   = (float*)   alloc((size_t)5 * 512 * 4);

  auto zero = [&](void* p, long long nf) {
    long long g = (nf + 255) / 256;
    k_zero<<<dim3((unsigned)g), dim3(256), 0, stream>>>((float*)p, nf);
  };

  auto gemm = [&](int mode, const void* X, const int* si, const int* di,
                  const float* W1p, const float* B1p, const float* W2p, const float* B2p,
                  float* Y, int rows, float* cs, float* cq) {
    dim3 g((rows + 127) / 128), b(256);
    if (mode == 0)
      k_gemm_chain<0><<<g, b, 0, stream>>>(X, si, di, W1p, B1p, W2p, B2p, Y, rows, cs, cq);
    else if (mode == 1)
      k_gemm_chain<1><<<g, b, 0, stream>>>(X, si, di, W1p, B1p, W2p, B2p, Y, rows, cs, cq);
    else
      k_gemm_chain<2><<<g, b, 0, stream>>>(X, si, di, W1p, B1p, W2p, B2p, Y, rows, cs, cq);
  };

  float* cs0 = stats + 0 * 512; float* cq0 = cs0 + 128; float* mi0 = cs0 + 256;
  float* cs1 = stats + 1 * 512; float* cq1 = cs1 + 128; float* mi1 = cs1 + 256;
  float* cs2 = stats + 2 * 512; float* cq2 = cs2 + 128; float* mi2 = cs2 + 256;
  float* cs3 = stats + 3 * 512; float* cq3 = cs3 + 128; float* mi3 = cs3 + 256;
  float* cs4 = stats + 4 * 512; float* cq4 = cs4 + 128; float* mi4 = cs4 + 256;

  // --- init scratch that is accumulated into ---
  zero(stats, 5 * 512);
  zero(umaxN, NN); zero(ssumN, NN); zero(cntN, NN); zero(accN, (long long)NN * 128);
  zero(umaxC, NCY); zero(ssumC, NCY); zero(cntC, NCY); zero(accC, (long long)NCY * 128);
  zero(gsum, (long long)GG * 128);

  // --- node embeddings (bf16) ---
  {
    long long n = (long long)NN * 64;
    k_embed<<<dim3((unsigned)((n + 255) / 256)), dim3(256), 0, stream>>>(sub_x, emb, x_bf, n);
  }

  // --- h = (concat @ w1 + b1) @ w2 + b2 ; BN stats ---
  gemm(1, x_bf, src, dst, w1, b1, w2, b2, bufA, EE, cs0, cq0);
  k_bn_fin<<<1, 128, 0, stream>>>(cs0, cq0, 1.0f / EE, mi0);
  {
    long long n = (long long)EE * 128;
    k_bn_act<<<dim3((unsigned)((n + 255) / 256)), dim3(256), 0, stream>>>(
        bufA, mi0, g0, beta0, nullptr, h_bf, n);
  }

  // --- bond branch: segment softmax over dst, weighted mean, GEMM chain ---
  k_att<<<dim3((EE + 7) / 8), dim3(256), 0, stream>>>(h_bf, nullptr, watt_b, att_e, EE);
  k_seg_max<<<dim3((EE + 255) / 256), dim3(256), 0, stream>>>(att_e, dst, umaxN, EE);
  k_seg_expsum<<<dim3((EE + 255) / 256), dim3(256), 0, stream>>>(att_e, dst, umaxN, ex_e, ssumN, cntN, EE);
  {
    long long n = (long long)EE * 128;
    k_seg_acc<<<dim3((unsigned)((n + 255) / 256)), dim3(256), 0, stream>>>(
        h_bf, nullptr, dst, ex_e, ssumN, accN, n);
  }
  {
    long long n = (long long)NN * 128;
    k_div_cnt<<<dim3((unsigned)((n + 255) / 256)), dim3(256), 0, stream>>>(accN, cntN, bond_bf, n);
  }
  gemm(0, bond_bf, nullptr, nullptr, wb1, nullptr, wb2, nullptr, Ybond, NN, cs1, cq1);
  k_bn_fin<<<1, 128, 0, stream>>>(cs1, cq1, 1.0f / NN, mi1);
  {
    long long n = (long long)NN * 128;
    k_bn_act<<<dim3((unsigned)((n + 255) / 256)), dim3(256), 0, stream>>>(
        Ybond, mi1, g1, beta1, bondF, nullptr, n);
  }

  // --- cycle branch ---
  k_att<<<dim3((MM + 7) / 8), dim3(256), 0, stream>>>(h_bf, cidx, watt_c, att_c, MM);
  k_seg_max<<<dim3((MM + 255) / 256), dim3(256), 0, stream>>>(att_c, cseg, umaxC, MM);
  k_seg_expsum<<<dim3((MM + 255) / 256), dim3(256), 0, stream>>>(att_c, cseg, umaxC, ex_c, ssumC, cntC, MM);
  {
    long long n = (long long)MM * 128;
    k_seg_acc<<<dim3((unsigned)((n + 255) / 256)), dim3(256), 0, stream>>>(
        h_bf, cidx, cseg, ex_c, ssumC, accC, n);
  }
  {
    long long n = (long long)NCY * 128;
    k_div_cnt<<<dim3((unsigned)((n + 255) / 256)), dim3(256), 0, stream>>>(accC, cntC, cyc_bf, n);
  }
  gemm(0, cyc_bf, nullptr, nullptr, wc1, nullptr, wc2, nullptr, Ycyc, NCY, cs2, cq2);
  k_bn_fin<<<1, 128, 0, stream>>>(cs2, cq2, 1.0f / NCY, mi2);
  {
    long long n = (long long)NCY * 128;
    k_bn_act<<<dim3((unsigned)((n + 255) / 256)), dim3(256), 0, stream>>>(
        Ycyc, mi2, g2, beta2, cycF, nullptr, n);
  }

  // --- e = h @ we1 @ we2 + cyc scatter + bond gather, then wl chain ---
  gemm(0, h_bf, nullptr, nullptr, we1, nullptr, we2, nullptr, e_buf, EE, nullptr, nullptr);
  {
    long long n = (long long)MM * 128;
    k_scatter_add<<<dim3((unsigned)((n + 255) / 256)), dim3(256), 0, stream>>>(
        cidx, cseg, cycF, e_buf, n);
  }
  {
    long long n = (long long)EE * 128;
    k_add_bond<<<dim3((unsigned)((n + 255) / 256)), dim3(256), 0, stream>>>(
        e_buf, bondF, src, dst, n);
  }
  gemm(2, e_buf, nullptr, nullptr, wl1, nullptr, wl2, nullptr, bufA, EE, cs3, cq3);
  k_bn_fin<<<1, 128, 0, stream>>>(cs3, cq3, 1.0f / EE, mi3);
  {
    long long n = (long long)EE * 128;
    k_bn_act<<<dim3((unsigned)((n + 255) / 256)), dim3(256), 0, stream>>>(
        bufA, mi3, g3, beta3, e_buf, nullptr, n);
  }

  // --- graph pooling + output head ---
  {
    long long n = (long long)EE * 128;
    k_graph_sum<<<dim3((unsigned)((n + 255) / 256)), dim3(256), 0, stream>>>(
        e_buf, src, batch, gsum, n);
  }
  gemm(2, gsum, nullptr, nullptr, wg1, nullptr, wg2, nullptr, Yg, GG, cs4, cq4);
  k_bn_fin<<<1, 128, 0, stream>>>(cs4, cq4, 1.0f / GG, mi4);
  {
    long long n = (long long)GG * 128;
    k_bn_act<<<dim3((unsigned)((n + 255) / 256)), dim3(256), 0, stream>>>(
        Yg, mi4, g4, beta4, gF, nullptr, n);
  }
  k_out<<<dim3((GG * 2 + 255) / 256), dim3(256), 0, stream>>>(gF, wout, (float*)d_out, GG);
}
